// LstmModel_87462714016314
// MI455X (gfx1250) — matrix operand order repaired and checked by reading, not compiled, not run on hardware
//
#include <hip/hip_runtime.h>
#include <hip/hip_bf16.h>

// Persistent single-WGP LSTM: 4 stacked LSTMCells, B=128, T=5000.
// One workgroup of 1024 threads (32 wave32s) owns the whole recurrence:
//  - f16 weights resident in LDS (converted once at kernel start)
//  - h states in LDS (f16, ping-pong), c states in VGPRs per wave
//  - GEMMs via v_wmma_f32_16x16x32_f16, fp32 accumulate
//  - gates via branch-free TRANS ops (v_exp/v_rcp, native v_tanh if available)
// Wave (m,j): m = batch tile (16 rows), j = h-column group (16 cols).
// Gate matrix [128 x 256]: wave (m,j) computes N-tiles {j, j+4, j+8, j+12}
// = i,f,g,o for its h-columns -> cell update is lane-local register math.

typedef __attribute__((ext_vector_type(16))) _Float16 v16h;
typedef __attribute__((ext_vector_type(8)))  _Float16 v8h;
typedef __attribute__((ext_vector_type(8)))  float    v8f;

#define BATCH 128
#define HD    64      // hidden width of layers 0..2
#define FH    256     // 4*HD gate width
#define BH    (BATCH*HD)

union Vh { v16h v; v8h h8[2]; };

// Branch-free sigmoid: rcp(1 + exp2(-log2(e)*x))  -> v_exp_f32 + v_rcp_f32
__device__ __forceinline__ float sigf(float x) {
    return __builtin_amdgcn_rcpf(1.0f + __builtin_amdgcn_exp2f(-1.44269504f * x));
}

// Native V_TANH_F32 on CDNA5 if the toolchain exposes it; else branch-free fallback
__device__ __forceinline__ float tanh_fast(float x) {
#if __has_builtin(__builtin_amdgcn_tanhf)
    return __builtin_amdgcn_tanhf(x);
#else
    // tanh(x) = 1 - 2/(1 + exp(2x)) = 1 - 2*rcp(1 + exp2(2*log2(e)*x))
    return 1.0f - 2.0f * __builtin_amdgcn_rcpf(
        1.0f + __builtin_amdgcn_exp2f(2.88539008f * x));
#endif
}

// A operand (16x32 f16 tile of h, row-major [rows][64] in LDS).
// ISA 7.12.2 layout: lanes 0-15 hold row M=lane, K = kb+{0..7, 16..23};
// lanes 16-31 hold row M=lane-16, K = kb+{8..15, 24..31}. Two ds_load_b128.
__device__ __forceinline__ v16h ldA(const _Float16* Abase, int row0, int kb, int lane) {
    const _Float16* p = Abase + (row0 + (lane & 15)) * HD + kb + ((lane >> 4) << 3);
    Vh u;
    u.h8[0] = *(const v8h*)(p);
    u.h8[1] = *(const v8h*)(p + 16);
    return u.v;
}

// B operand (32x16 f16 tile of W^T). We keep W row-major [256 n][64 k] in LDS
// (exactly the input layout), matching the A K order:
// lane L holds col N = L%16, K = kb + (L/16)*8 + {0..7, 16..23}. Two ds_load_b128.
__device__ __forceinline__ v16h ldB(const _Float16* Wbase, int n0, int kb, int lane) {
    const _Float16* p = Wbase + (n0 + (lane & 15)) * HD + kb + ((lane >> 4) << 3);
    Vh u;
    u.h8[0] = *(const v8h*)(p);
    u.h8[1] = *(const v8h*)(p + 16);
    return u.v;
}

// Accumulate gates += A[rows of batch tile] x W^T into 4 gate tiles (i,f,g,o).
__device__ __forceinline__ void gemm_acc(const _Float16* A, const _Float16* W,
                                         int row0, int j, int lane, v8f acc[4]) {
#pragma unroll
    for (int kb = 0; kb < HD; kb += 32) {
        v16h a = ldA(A, row0, kb, lane);
#pragma unroll
        for (int g = 0; g < 4; ++g) {
            v16h b = ldB(W, g * HD + j * 16, kb, lane);
            acc[g] = __builtin_amdgcn_wmma_f32_16x16x32_f16(
                false, a, false, b, (short)0, acc[g], false, false);
        }
    }
}

__device__ __forceinline__ void bias_init(const float* Bs, int j, int lane, v8f acc[4]) {
#pragma unroll
    for (int g = 0; g < 4; ++g) {
        float bn = Bs[g * HD + j * 16 + (lane & 15)];
#pragma unroll
        for (int r = 0; r < 8; ++r) acc[g][r] = bn;
    }
}

// LSTM cell update on register tiles; writes h tile (f16) to LDS.
// D-tile layout (ISA 7.12.2): acc[r] holds row M = r + 8*(lane>=16), col N = lane%16.
__device__ __forceinline__ void cell_update(v8f acc[4], v8f& c, _Float16* Hout,
                                            int row0, int j, int lane) {
#pragma unroll
    for (int r = 0; r < 8; ++r) {
        float ci = sigf(acc[0][r]);
        float cf = sigf(acc[1][r]);
        float cg = tanh_fast(acc[2][r]);
        float co = sigf(acc[3][r]);
        float cn = cf * c[r] + ci * cg;
        c[r] = cn;
        float h = co * tanh_fast(cn);
        int row = row0 + r + ((lane >> 4) << 3);
        Hout[row * HD + j * 16 + (lane & 15)] = (_Float16)h;
    }
}

__global__ __launch_bounds__(1024, 1) void lstm_persistent_kernel(
    const float* __restrict__ x,
    const float* __restrict__ Wih0, const float* __restrict__ Whh0,
    const float* __restrict__ bih0, const float* __restrict__ bhh0,
    const float* __restrict__ Wih1, const float* __restrict__ Whh1,
    const float* __restrict__ bih1, const float* __restrict__ bhh1,
    const float* __restrict__ Wih2, const float* __restrict__ Whh2,
    const float* __restrict__ bih2, const float* __restrict__ bhh2,
    const float* __restrict__ Wih3, const float* __restrict__ Whh3,
    const float* __restrict__ bih3, const float* __restrict__ bhh3,
    float* __restrict__ out, int T) {
    // ---- LDS (about 272 KB of the WGP's 320 KB) ----
    __shared__ __align__(16) _Float16 sWhh0[FH * HD];
    __shared__ __align__(16) _Float16 sWih1[FH * HD];
    __shared__ __align__(16) _Float16 sWhh1[FH * HD];
    __shared__ __align__(16) _Float16 sWih2[FH * HD];
    __shared__ __align__(16) _Float16 sWhh2[FH * HD];
    __shared__ __align__(16) _Float16 sWih3[16 * HD];     // 4x64 padded to 16x64
    __shared__ __align__(16) _Float16 sH0[2 * BH];        // ping-pong h states
    __shared__ __align__(16) _Float16 sH1[2 * BH];
    __shared__ __align__(16) _Float16 sH2[2 * BH];
    __shared__ __align__(16) float sWih0f[FH];
    __shared__ __align__(16) float sWhh3f[16];
    __shared__ __align__(16) float sB0[FH], sB1[FH], sB2[FH], sB3[16];
    __shared__ __align__(16) float sU[BATCH];             // per-batch scalar input this step
    __shared__ __align__(16) float sY[BATCH];             // previous output h3
    __shared__ __align__(16) float sC3[BATCH];            // layer-3 cell state
    __shared__ __align__(16) float sG3[8 * 16 * 16];      // layer-3 gate tiles scratch

    const int tid  = threadIdx.x;
    const int lane = tid & 31;
    const int wave = tid >> 5;
    const int m    = wave >> 2;        // batch tile 0..7
    const int j    = wave & 3;         // h-column group 0..3
    const int row0 = m * 16;

    // ---- one-time: convert weights to f16 into LDS, fold biases, zero state ----
    for (int i = tid; i < FH * HD; i += 1024) {
        sWhh0[i] = (_Float16)Whh0[i];
        sWih1[i] = (_Float16)Wih1[i];
        sWhh1[i] = (_Float16)Whh1[i];
        sWih2[i] = (_Float16)Wih2[i];
        sWhh2[i] = (_Float16)Whh2[i];
    }
    for (int i = tid; i < 16 * HD; i += 1024)
        sWih3[i] = (i < 4 * HD) ? (_Float16)Wih3[i] : (_Float16)0.0f;
    for (int i = tid; i < FH; i += 1024) {
        sWih0f[i] = Wih0[i];                 // Wih0 is (256,1)
        sB0[i] = bih0[i] + bhh0[i];
        sB1[i] = bih1[i] + bhh1[i];
        sB2[i] = bih2[i] + bhh2[i];
    }
    if (tid < 16) { sWhh3f[tid] = (tid < 4) ? Whh3[tid] : 0.0f;
                    sB3[tid]    = (tid < 4) ? (bih3[tid] + bhh3[tid]) : 0.0f; }
    for (int i = tid; i < 2 * BH; i += 1024) {
        sH0[i] = (_Float16)0.0f; sH1[i] = (_Float16)0.0f; sH2[i] = (_Float16)0.0f;
    }
    if (tid < BATCH) { sU[tid] = 0.0f; sY[tid] = 0.0f; sC3[tid] = 0.0f; }
    __syncthreads();

    // c states for layers 0..2 live in VGPRs for the entire scan
    v8f c0t = {}, c1t = {}, c2t = {};

    for (int t = 0; t < T; ++t) {
        const int cur = t & 1, prv = cur ^ 1;
        _Float16* h0c = sH0 + cur * BH; const _Float16* h0p = sH0 + prv * BH;
        _Float16* h1c = sH1 + cur * BH; const _Float16* h1p = sH1 + prv * BH;
        _Float16* h2c = sH2 + cur * BH; const _Float16* h2p = sH2 + prv * BH;

        // issue next-step input load early; consumed after all 4 layers
        float xr = 0.0f;
        if (tid < BATCH) {
            xr = x[(size_t)tid * T + t];                       // delayed index (t+1)-1
            if (t + 16 < T) __builtin_prefetch(&x[(size_t)tid * T + t + 16], 0, 1);
        }

        // ---- layer 0: gates = u*Wih0 + h0_prev @ Whh0^T + bias ----
        {
            float uv[8];
#pragma unroll
            for (int r = 0; r < 8; ++r)
                uv[r] = sU[row0 + r + ((lane >> 4) << 3)];
            v8f acc[4];
#pragma unroll
            for (int g = 0; g < 4; ++g) {
                int n = g * HD + j * 16 + (lane & 15);
                float bn = sB0[n], w0 = sWih0f[n];
#pragma unroll
                for (int r = 0; r < 8; ++r)
                    acc[g][r] = fmaf(uv[r], w0, bn);
            }
            gemm_acc(h0p, sWhh0, row0, j, lane, acc);
            cell_update(acc, c0t, h0c, row0, j, lane);
        }
        __syncthreads();

        // ---- layer 1 ----
        {
            v8f acc[4];
            bias_init(sB1, j, lane, acc);
            gemm_acc(h0c, sWih1, row0, j, lane, acc);
            gemm_acc(h1p, sWhh1, row0, j, lane, acc);
            cell_update(acc, c1t, h1c, row0, j, lane);
        }
        __syncthreads();

        // ---- layer 2 ----
        {
            v8f acc[4];
            bias_init(sB2, j, lane, acc);
            gemm_acc(h1c, sWih2, row0, j, lane, acc);
            gemm_acc(h2p, sWhh2, row0, j, lane, acc);
            cell_update(acc, c2t, h2c, row0, j, lane);
        }
        __syncthreads();

        // ---- layer 3: 128x64x16 WMMA (only 4 of 16 N-cols used), 8 waves ----
        if ((wave & 3) == 0) {                 // wave-uniform branch, EXEC stays all-1s
            v8f acc3 = {};
#pragma unroll
            for (int kb = 0; kb < HD; kb += 32) {
                v16h a = ldA(h2c, row0, kb, lane);
                v16h b = ldB(sWih3, 0, kb, lane);
                acc3 = __builtin_amdgcn_wmma_f32_16x16x32_f16(
                    false, a, false, b, (short)0, acc3, false, false);
            }
#pragma unroll
            for (int r = 0; r < 8; ++r) {
                int row = r + ((lane >> 4) << 3);
                sG3[m * 256 + row * 16 + (lane & 15)] = acc3[r];
            }
        }
        __syncthreads();

        // ---- layer-3 scalar tail + output + next-step input ----
        if (tid < BATCH) {
            const int b = tid;
            const float* gt = &sG3[(b >> 4) * 256 + (b & 15) * 16];
            float yp = sY[b];
            float gi = gt[0] + sB3[0] + yp * sWhh3f[0];
            float gf = gt[1] + sB3[1] + yp * sWhh3f[1];
            float gg = gt[2] + sB3[2] + yp * sWhh3f[2];
            float go = gt[3] + sB3[3] + yp * sWhh3f[3];
            float cn = sigf(gf) * sC3[b] + sigf(gi) * tanh_fast(gg);
            float h3 = sigf(go) * tanh_fast(cn);
            sC3[b] = cn;
            sY[b]  = h3;
            sU[b]  = xr - h3;                  // realistic_hindsight feedback
            out[(size_t)b * T + t] = h3;
        }
        __syncthreads();
    }
}

extern "C" void kernel_launch(void* const* d_in, const int* in_sizes, int n_in,
                              void* d_out, int out_size, void* d_ws, size_t ws_size,
                              hipStream_t stream) {
    const float* x    = (const float*)d_in[0];
    const float* Wih0 = (const float*)d_in[1];
    const float* Whh0 = (const float*)d_in[2];
    const float* bih0 = (const float*)d_in[3];
    const float* bhh0 = (const float*)d_in[4];
    const float* Wih1 = (const float*)d_in[5];
    const float* Whh1 = (const float*)d_in[6];
    const float* bih1 = (const float*)d_in[7];
    const float* bhh1 = (const float*)d_in[8];
    const float* Wih2 = (const float*)d_in[9];
    const float* Whh2 = (const float*)d_in[10];
    const float* bih2 = (const float*)d_in[11];
    const float* bhh2 = (const float*)d_in[12];
    const float* Wih3 = (const float*)d_in[13];
    const float* Whh3 = (const float*)d_in[14];
    const float* bih3 = (const float*)d_in[15];
    const float* bhh3 = (const float*)d_in[16];
    float* out = (float*)d_out;

    const int T = in_sizes[0] / BATCH;   // x is [B=128, C=1, T]

    lstm_persistent_kernel<<<dim3(1), dim3(1024), 0, stream>>>(
        x, Wih0, Whh0, bih0, bhh0, Wih1, Whh1, bih1, bhh1,
        Wih2, Whh2, bih2, bhh2, Wih3, Whh3, bih3, bhh3, out, T);
}
